// DeeperGCN_1039382086077
// MI455X (gfx1250) — matrix operand order repaired and checked by reading, not compiled, not run on hardware
//
#include <hip/hip_runtime.h>
#include <math.h>

#define NN 50000
#define NE 800000
#define NG 256
#define H 128
#define LL 8
#define TT 2
#define NODE_IN 64
#define EDGE_IN 16
#define EPSF 1e-7f
#define LN_EPSF 1e-5f

typedef __attribute__((ext_vector_type(16))) _Float16 v16h;
typedef __attribute__((ext_vector_type(8)))  _Float16 v8h;
typedef __attribute__((ext_vector_type(8)))  float    v8f;

// ---- order-preserving float<->uint key (monotonic under unsigned compare) ----
__device__ __forceinline__ unsigned f2key(float x) {
    unsigned b = __float_as_uint(x);
    return b ^ (unsigned)((((int)b) >> 31) | 0x80000000);
}
__device__ __forceinline__ float key2f(unsigned k) {
    unsigned b = k ^ ((k & 0x80000000u) ? 0x80000000u : 0xFFFFFFFFu);
    return __uint_as_float(b);
}

// ---------------- pack f32 weights [K,Nout] into WMMA B-fragment order (f16) ----------------
// layout: for column-tile ct (16 cols), k-step ks (K=32 slab), lane l (32), 16 halves contig:
//   lane holds column ct*16+(l&15), K indices ks*32 + (l>>4)*16 + {0..15}
__global__ void pack_b_kernel(const float* __restrict__ B, _Float16* __restrict__ out,
                              int K, int Nout)
{
    int idx = blockIdx.x * 256 + threadIdx.x;
    if (idx >= K * Nout) return;
    int v    = idx & 15;
    int lane = (idx >> 4) & 31;
    int t    = idx >> 9;            // ct*nK + ks
    int nK   = K >> 5;
    int ks   = t % nK;
    int ct   = t / nK;
    int col  = ct * 16 + (lane & 15);
    int kk   = ks * 32 + ((lane >> 4) << 4) + v;
    out[idx] = (_Float16)B[(long)kk * Nout + col];
}

__global__ void zero_f32_kernel(float* __restrict__ p, long n) {
    long i = (long)blockIdx.x * blockDim.x + threadIdx.x;
    if (i < n) p[i] = 0.0f;
}

// ---------------- WMMA GEMM: C[M,Nout] = A[M,K](f32) * Bpacked(f16) + bias (+resid)(+relu) ----
// grid.x = M/16 tiles, grid.y = Nout/128 tiles, block = 256 (8 wave32 waves)
template<int K, bool HAS_RES, bool RELU>
__global__ __launch_bounds__(256)
void gemm_wmma_kernel(const float* __restrict__ A,
                      const _Float16* __restrict__ Bp, int Nout,
                      const float* __restrict__ bias,
                      const float* __restrict__ resid,
                      float* __restrict__ C)
{
    constexpr int LDA = K + 8;                 // +8 halves: rows offset by 4 LDS banks
    __shared__ _Float16 As[16 * LDA];
    const int tid  = threadIdx.x;
    const int mb   = blockIdx.x * 16;
    const int wave = tid >> 5;
    const int lane = tid & 31;
    const int ct   = blockIdx.y * 8 + wave;    // global 16-column tile index
    constexpr int nK = K >> 5;

    // stage & convert A tile [16,K] into LDS as f16 (padded rows)
#pragma unroll
    for (int i = tid; i < 16 * K; i += 256) {
        int r = i / K, k = i % K;
        As[r * LDA + k] = (_Float16)A[(long)(mb + r) * K + k];
    }
    __syncthreads();

    v8f acc = {};
    const int row   = lane & 15;               // A: M = lane%16
    const int koffA = (lane >> 4) * 16;        // A: half-wave K phase (contiguous, matches packed B)
    const int col   = lane & 15;               // B/C: N = lane%16

    const _Float16* bbase = Bp + (((long)ct * nK * 32 + lane) << 4);
#pragma unroll
    for (int ks = 0; ks < nK; ++ks) {
        const int k0 = ks * 32;
        v16h a;
#pragma unroll
        for (int j = 0; j < 8; ++j) {          // V0-7 -> K+{0..15|16..31} per half-wave
            int kk = k0 + koffA + 2 * j;
            a[2 * j]     = As[row * LDA + kk];
            a[2 * j + 1] = As[row * LDA + kk + 1];
        }
        const _Float16* bptr = bbase + (long)ks * 512;   // 32 lanes * 16 halves
        v8h blo = *(const v8h*)(bptr);
        v8h bhi = *(const v8h*)(bptr + 8);
        v16h b = __builtin_shufflevector(blo, bhi, 0, 1, 2, 3, 4, 5, 6, 7,
                                         8, 9, 10, 11, 12, 13, 14, 15);
        acc = __builtin_amdgcn_wmma_f32_16x16x32_f16(false, a, false, b, (short)0, acc,
                                                     false, false);
    }

    const int ro   = (lane >> 4) * 8;          // C/D: VGPR r -> M=r | M=r+8 per half-wave
    const int gcol = ct * 16 + col;
    const float bi = bias[gcol];
#pragma unroll
    for (int r = 0; r < 8; ++r) {
        int grow = mb + ro + r;
        float v = acc[r] + bi;
        if (HAS_RES) v += resid[(long)grow * Nout + gcol];
        if (RELU)    v = fmaxf(v, 0.0f);
        C[(long)grow * Nout + gcol] = v;
    }
}

// ---------------- LayerNorm + ReLU, and reset per-layer reduction buffers ----------------
__global__ __launch_bounds__(128)
void ln_relu_init_kernel(const float* __restrict__ hv,
                         const float* __restrict__ sc, const float* __restrict__ bi,
                         float* __restrict__ h1, unsigned* __restrict__ mxkey,
                         float* __restrict__ den, float* __restrict__ num)
{
    __shared__ float s1[128], s2[128];
    const int n = blockIdx.x, h = threadIdx.x;
    const long idx = (long)n * H + h;
    float x = hv[idx];
    s1[h] = x; s2[h] = x * x;
    __syncthreads();
    for (int s = 64; s > 0; s >>= 1) {
        if (h < s) { s1[h] += s1[h + s]; s2[h] += s2[h + s]; }
        __syncthreads();
    }
    float mu  = s1[0] * (1.0f / H);
    float var = s2[0] * (1.0f / H) - mu * mu;
    float y = (x - mu) * rsqrtf(var + LN_EPSF) * sc[h] + bi[h];
    h1[idx]    = fmaxf(y, 0.0f);
    mxkey[idx] = f2key(-3.0e38f);
    den[idx]   = 0.0f;
    num[idx]   = 0.0f;
}

// ---------------- edge pass 1: per-(dst,channel) max of beta*m ----------------
__global__ __launch_bounds__(128)
void edge_max_kernel(const float* __restrict__ ef, const int* __restrict__ src,
                     const int* __restrict__ dst,
                     const float* __restrict__ Wee, const float* __restrict__ bee,
                     const float* __restrict__ h1, const float* __restrict__ betas, int l,
                     unsigned* __restrict__ mxkey)
{
    __shared__ float sef[EDGE_IN];
    __shared__ int   ssd[2];
    const long e = blockIdx.x;
    const int  h = threadIdx.x;
    if (h < EDGE_IN) sef[h] = ef[e * EDGE_IN + h];
    if (h == 0) { ssd[0] = src[e]; ssd[1] = dst[e]; }
    __syncthreads();
    float he = bee[h];
#pragma unroll
    for (int j = 0; j < EDGE_IN; ++j) he = fmaf(sef[j], Wee[j * H + h], he);
    float m = fmaxf(h1[(long)ssd[0] * H + h] + he, 0.0f) + EPSF;
    atomicMax(&mxkey[(long)ssd[1] * H + h], f2key(betas[l] * m));
}

// ---------------- edge pass 2: fused exp + denominator + weighted numerator ----------------
__global__ __launch_bounds__(128)
void edge_acc_kernel(const float* __restrict__ ef, const int* __restrict__ src,
                     const int* __restrict__ dst,
                     const float* __restrict__ Wee, const float* __restrict__ bee,
                     const float* __restrict__ h1, const float* __restrict__ betas, int l,
                     const unsigned* __restrict__ mxkey,
                     float* __restrict__ den, float* __restrict__ num)
{
    __shared__ float sef[EDGE_IN];
    __shared__ int   ssd[2];
    const long e = blockIdx.x;
    const int  h = threadIdx.x;
    if (h < EDGE_IN) sef[h] = ef[e * EDGE_IN + h];
    if (h == 0) { ssd[0] = src[e]; ssd[1] = dst[e]; }
    __syncthreads();
    float he = bee[h];
#pragma unroll
    for (int j = 0; j < EDGE_IN; ++j) he = fmaf(sef[j], Wee[j * H + h], he);
    float m  = fmaxf(h1[(long)ssd[0] * H + h] + he, 0.0f) + EPSF;
    const long di = (long)ssd[1] * H + h;
    float ex = __expf(betas[l] * m - key2f(mxkey[di]));
    atomicAdd(&den[di], ex);
    atomicAdd(&num[di], ex * m);
}

// ---------------- x = h1 + num/den ----------------
__global__ void combine_kernel(const float* __restrict__ h1, const float* __restrict__ num,
                               const float* __restrict__ den, float* __restrict__ xb)
{
    long i = (long)blockIdx.x * blockDim.x + threadIdx.x;
    if (i < (long)NN * H) {
        float d = den[i];
        xb[i] = h1[i] + (d > 0.0f ? num[i] / d : 0.0f);
    }
}

// ---------------- readout: gf = segment_sum(hv) ----------------
__global__ __launch_bounds__(128)
void gf_acc_kernel(const float* __restrict__ hv, const int* __restrict__ gid,
                   float* __restrict__ gf)
{
    const int n = blockIdx.x, h = threadIdx.x;
    atomicAdd(&gf[(long)gid[n] * H + h], hv[(long)n * H + h]);
}

// ---------------- attention logit z[n] + per-graph max ----------------
__global__ __launch_bounds__(128)
void attn_logit_kernel(const float* __restrict__ gf, const float* __restrict__ hv,
                       const int* __restrict__ gid,
                       const float* __restrict__ logW, const float* __restrict__ logB,
                       float* __restrict__ z, unsigned* __restrict__ zmaxkey)
{
    __shared__ float s1[128];
    const int n = blockIdx.x, h = threadIdx.x;
    const int g = gid[n];
    float t = fmaxf(gf[(long)g * H + h], 0.0f) * logW[h] + hv[(long)n * H + h] * logW[H + h];
    s1[h] = t;
    __syncthreads();
    for (int s = 64; s > 0; s >>= 1) {
        if (h < s) s1[h] += s1[h + s];
        __syncthreads();
    }
    if (h == 0) {
        float zz = s1[0] + logB[0];
        zz = (zz > 0.0f) ? zz : 0.01f * zz;      // leaky_relu(0.01)
        z[n] = zz;
        atomicMax(&zmaxkey[g], f2key(zz));
    }
}

// ---------------- ez accumulation: ezsum[g], ctxacc[g,h] += ez*proj ----------------
__global__ __launch_bounds__(128)
void ctx_acc_kernel(const float* __restrict__ z, const unsigned* __restrict__ zmaxkey,
                    const int* __restrict__ gid, const float* __restrict__ proj,
                    float* __restrict__ ezsum, float* __restrict__ ctxacc)
{
    const int n = blockIdx.x, h = threadIdx.x;
    const int g = gid[n];
    float ez = __expf(z[n] - key2f(zmaxkey[g]));
    if (h == 0) atomicAdd(&ezsum[g], ez);
    atomicAdd(&ctxacc[(long)g * H + h], ez * proj[(long)n * H + h]);
}

// ---------------- ctx = elu(ctxacc / ezsum) ----------------
__global__ void ctx_elu_kernel(const float* __restrict__ ctxacc, const float* __restrict__ ezsum,
                               float* __restrict__ ctx)
{
    int i = blockIdx.x * 128 + threadIdx.x;   // i < G*H
    int g = i >> 7;
    float s = ezsum[g];
    float c = (s > 0.0f) ? ctxacc[i] / s : 0.0f;
    ctx[i] = (c > 0.0f) ? c : expm1f(c);
}

// ---------------- GRU matvecs: gi = Wih*ctx + bih ; gh = Whh*gf + bhh ----------------
__global__ __launch_bounds__(128)
void gru_mv_kernel(const float* __restrict__ ctx, const float* __restrict__ gf,
                   const float* __restrict__ Wih, const float* __restrict__ Whh,
                   const float* __restrict__ bih, const float* __restrict__ bhh,
                   float* __restrict__ gi, float* __restrict__ gh)
{
    const int g = blockIdx.x / 3;
    const int i = (blockIdx.x % 3) * 128 + threadIdx.x;   // 0..383
    float ai = bih[i], ah = bhh[i];
    const float* wi = Wih + (long)i * H;
    const float* wh = Whh + (long)i * H;
    const float* c  = ctx + (long)g * H;
    const float* f  = gf  + (long)g * H;
    for (int k = 0; k < H; ++k) {
        ai = fmaf(wi[k], c[k], ai);
        ah = fmaf(wh[k], f[k], ah);
    }
    gi[(long)g * 384 + i] = ai;
    gh[(long)g * 384 + i] = ah;
}

// ---------------- GRU gates + relu, gf updated in place ----------------
__global__ void gru_gate_kernel(const float* __restrict__ gi, const float* __restrict__ gh,
                                float* __restrict__ gf)
{
    int idx = blockIdx.x * 128 + threadIdx.x;   // G*H
    int g = idx >> 7, h = idx & 127;
    const float* a = gi + (long)g * 384;
    const float* b = gh + (long)g * 384;
    float r  = 1.0f / (1.0f + __expf(-(a[h] + b[h])));
    float zg = 1.0f / (1.0f + __expf(-(a[h + 128] + b[h + 128])));
    float nn = tanhf(a[h + 256] + r * b[h + 256]);
    float o  = (1.0f - zg) * nn + zg * gf[idx];
    gf[idx] = fmaxf(o, 0.0f);
}

// ---------------- final: out[g] = hidden[g,:] . W2 + b2 ----------------
__global__ __launch_bounds__(128)
void final_out_kernel(const float* __restrict__ hidden, const float* __restrict__ W2,
                      const float* __restrict__ b2, float* __restrict__ out)
{
    __shared__ float s1[128];
    const int g = blockIdx.x, h = threadIdx.x;
    float acc = 0.0f;
    for (int j = h; j < 1024; j += 128) acc = fmaf(hidden[(long)g * 1024 + j], W2[j], acc);
    s1[h] = acc;
    __syncthreads();
    for (int s = 64; s > 0; s >>= 1) {
        if (h < s) s1[h] += s1[h + s];
        __syncthreads();
    }
    if (h == 0) out[g] = s1[0] + b2[0];
}

// =======================================================================
extern "C" void kernel_launch(void* const* d_in, const int* in_sizes, int n_in,
                              void* d_out, int out_size, void* d_ws, size_t ws_size,
                              hipStream_t stream)
{
    (void)in_sizes; (void)n_in; (void)out_size; (void)ws_size;
    const float* node_feat = (const float*)d_in[0];
    const float* edge_feat = (const float*)d_in[1];
    const int*   src       = (const int*)  d_in[2];
    const int*   dst       = (const int*)  d_in[3];
    const int*   gid       = (const int*)  d_in[4];
    const float* W_ne      = (const float*)d_in[5];
    const float* b_ne      = (const float*)d_in[6];
    const float* W_ee      = (const float*)d_in[7];
    const float* b_ee      = (const float*)d_in[8];
    const float* ln_scale  = (const float*)d_in[9];
    const float* ln_bias   = (const float*)d_in[10];
    const float* betas     = (const float*)d_in[11];
    const float* mlp_W     = (const float*)d_in[12];
    const float* mlp_b     = (const float*)d_in[13];
    const float* logit_W   = (const float*)d_in[14];
    const float* logit_b   = (const float*)d_in[15];
    const float* proj_W    = (const float*)d_in[16];
    const float* proj_b    = (const float*)d_in[17];
    const float* gru_Wih   = (const float*)d_in[18];
    const float* gru_Whh   = (const float*)d_in[19];
    const float* gru_bih   = (const float*)d_in[20];
    const float* gru_bhh   = (const float*)d_in[21];
    const float* W1        = (const float*)d_in[22];
    const float* b1        = (const float*)d_in[23];
    const float* W2        = (const float*)d_in[24];
    const float* b2        = (const float*)d_in[25];
    float* out = (float*)d_out;

    // ---- bump allocator over d_ws ----
    char* p = (char*)d_ws;
    auto alloc = [&](size_t bytes) -> void* {
        void* r = (void*)p;
        p += (bytes + 255) & ~(size_t)255;
        return r;
    };
    const long NH = (long)NN * H;
    float*    hvA   = (float*)   alloc(NH * 4);
    float*    hvB   = (float*)   alloc(NH * 4);
    float*    h1    = (float*)   alloc(NH * 4);
    float*    xbuf  = (float*)   alloc(NH * 4);     // reused as proj in readout
    unsigned* mxkey = (unsigned*)alloc(NH * 4);
    float*    den   = (float*)   alloc(NH * 4);     // reused: z lives in den[0..NN)
    float*    num   = (float*)   alloc(NH * 4);
    float*    gf    = (float*)   alloc((long)NG * H * 4);
    float*    ctxac = (float*)   alloc((long)NG * H * 4);
    float*    ctx   = (float*)   alloc((long)NG * H * 4);
    float*    gi    = (float*)   alloc((long)NG * 384 * 4);
    float*    gh    = (float*)   alloc((long)NG * 384 * 4);
    float*    ezsum = (float*)   alloc((long)NG * 4);
    unsigned* zmk   = (unsigned*)alloc((long)NG * 4);
    float*    hid   = (float*)   alloc((long)NG * 1024 * 4);
    _Float16* wne16  = (_Float16*)alloc((long)NODE_IN * H * 2);
    _Float16* mlp16  = (_Float16*)alloc((long)LL * H * H * 2);
    _Float16* proj16 = (_Float16*)alloc((long)TT * H * H * 2);
    _Float16* w1_16  = (_Float16*)alloc((long)H * 1024 * 2);

    // ---- pre-pack weights into WMMA B-fragment order (f16) ----
    auto pack = [&](const float* in, _Float16* o, int K, int Nout) {
        pack_b_kernel<<<(unsigned)((K * Nout + 255) / 256), 256, 0, stream>>>(in, o, K, Nout);
    };
    pack(W_ne, wne16, NODE_IN, H);
    for (int l = 0; l < LL; ++l)
        pack(mlp_W + (long)l * H * H, mlp16 + (long)l * H * H, H, H);
    for (int t = 0; t < TT; ++t)
        pack(proj_W + (long)t * H * H, proj16 + (long)t * H * H, H, H);
    pack(W1, w1_16, H, 1024);

    // ---- node embedding: hvA = node_feat @ W_ne + b_ne  (WMMA, K=64) ----
    gemm_wmma_kernel<NODE_IN, false, false><<<dim3(NN / 16, 1), 256, 0, stream>>>(
        node_feat, wne16, H, b_ne, nullptr, hvA);

    // ---- 8 GENConv layers ----
    float* hv  = hvA;
    float* hvn = hvB;
    for (int l = 0; l < LL; ++l) {
        ln_relu_init_kernel<<<NN, 128, 0, stream>>>(
            hv, ln_scale + (long)l * H, ln_bias + (long)l * H, h1, mxkey, den, num);
        edge_max_kernel<<<NE, 128, 0, stream>>>(
            edge_feat, src, dst, W_ee, b_ee, h1, betas, l, mxkey);
        edge_acc_kernel<<<NE, 128, 0, stream>>>(
            edge_feat, src, dst, W_ee, b_ee, h1, betas, l, mxkey, den, num);
        combine_kernel<<<(unsigned)(NH / 256), 256, 0, stream>>>(h1, num, den, xbuf);
        gemm_wmma_kernel<H, true, false><<<dim3(NN / 16, 1), 256, 0, stream>>>(
            xbuf, mlp16 + (long)l * H * H, H, mlp_b + (long)l * H, hv, hvn);
        float* t = hv; hv = hvn; hvn = t;
    }

    // ---- readout init: gf = segment_sum(hv) ----
    zero_f32_kernel<<<(NG * H + 255) / 256, 256, 0, stream>>>(gf, (long)NG * H);
    gf_acc_kernel<<<NN, 128, 0, stream>>>(hv, gid, gf);

    float* z    = den;    // reuse
    float* proj = xbuf;   // reuse
    for (int t = 0; t < TT; ++t) {
        zero_f32_kernel<<<(NG * H + 255) / 256, 256, 0, stream>>>(ctxac, (long)NG * H);
        zero_f32_kernel<<<1, 256, 0, stream>>>(ezsum, NG);
        zero_f32_kernel<<<1, 256, 0, stream>>>((float*)zmk, NG);   // key 0 == -inf
        attn_logit_kernel<<<NN, 128, 0, stream>>>(
            gf, hv, gid, logit_W + (long)t * 2 * H, logit_b + t, z, zmk);
        gemm_wmma_kernel<H, false, false><<<dim3(NN / 16, 1), 256, 0, stream>>>(
            hv, proj16 + (long)t * H * H, H, proj_b + (long)t * H, nullptr, proj);
        ctx_acc_kernel<<<NN, 128, 0, stream>>>(z, zmk, gid, proj, ezsum, ctxac);
        ctx_elu_kernel<<<NG * H / 128, 128, 0, stream>>>(ctxac, ezsum, ctx);
        gru_mv_kernel<<<NG * 3, 128, 0, stream>>>(
            ctx, gf, gru_Wih + (long)t * 384 * H, gru_Whh + (long)t * 384 * H,
            gru_bih + (long)t * 384, gru_bhh + (long)t * 384, gi, gh);
        gru_gate_kernel<<<NG * H / 128, 128, 0, stream>>>(gi, gh, gf);
    }

    // ---- final MLP: hidden = relu(gf @ W1 + b1) (WMMA 16x8 tiles), out = hidden @ W2 + b2 ----
    gemm_wmma_kernel<H, false, true><<<dim3(NG / 16, 1024 / 128), 256, 0, stream>>>(
        gf, w1_16, 1024, b1, nullptr, hid);
    final_out_kernel<<<NG, 128, 0, stream>>>(hid, W2, b2, out);
}
